// GraphTrajectoryLSTM_49323404427369
// MI455X (gfx1250) — matrix operand order repaired and checked by reading, not compiled, not run on hardware
//
#include <hip/hip_runtime.h>
#include <hip/hip_bf16.h>
#include <math.h>

typedef __attribute__((ext_vector_type(16))) __bf16 v16bf;
typedef __attribute__((ext_vector_type(8)))  __bf16 v8bf;
typedef __attribute__((ext_vector_type(8)))  float  v8f;

#define DEV __device__ __forceinline__

// ---------------- problem constants ----------------
#define BB   128          // batch
#define NN   200          // nodes
#define NP   208          // nodes padded to 16
#define KP   224          // node contraction padded to 32
#define FF   16           // node features
#define TT   50           // input seq len
#define HH   512          // hidden
#define G4   2048         // 4*H
#define TOUT 30
#define MBN  (BB*NP)      // 26624 (multiple of 16)
#define KIN0 544          // lstm layer-0 K (513/514 -> 544, multiple of 32)
#define BT   (BB*TT)      // 6400

// =====================================================================
// Generic bf16 WMMA GEMM:  C[z] = alpha * A[z] @ B[z] (+bias)(+Add)(relu)
// A: [M,K] row-major bf16 ; Bt: [N,K] row-major bf16 (i.e. B transposed)
// Per-wave tile: 16 (M) x 64 (N), K step 32 via v_wmma_f32_16x16x32_bf16.
// Hot path (jmax==4) is branch-free and software-pipelined (loads for
// step k+32 issued before the 4 WMMAs of step k).
// Fragment layouts used here (contiguous k order, same for A and B):
//   A 16x32: lane<16 -> row=lane, k 0..15 contiguous; lane>=16 -> k 16..31
//   B 32x16: lane<16 -> col=lane, k 0..15 contiguous; lane>=16 -> k 16..31
//   C 16x16 f32: vgpr i -> row = i + (lane>=16 ? 8:0), col = lane&15
// =====================================================================
struct GemmP {
  const __bf16* A; const __bf16* Bt; const __bf16* Add; const float* bias;
  float* Cf; __bf16* Cb;
  int M, N, K, lda, ldb, ldc, ldadd, inner, tiles_m, ngroups, relu;
  long long sAo, sAi, sBo, sBi, sCo, sCi, sAddo, sBiaso;
  float alpha;
};

DEV v16bf ldA(const __bf16* ap) {
  union { v16bf v; v8bf h[2]; } a;
  a.h[0] = *(const v8bf*)ap;
  a.h[1] = *(const v8bf*)(ap + 8);
  return a.v;
}
DEV v16bf ldB(const __bf16* bp) { return *(const v16bf*)bp; }

#define WMMA_BF16(A_, B_, C_) \
  __builtin_amdgcn_wmma_f32_16x16x32_bf16(false, (A_), false, (B_), (short)0, (C_), false, false)

__global__ __launch_bounds__(256) void wmma_gemm_bf16(GemmP p) {
  const int wave = (int)((blockIdx.x * 256u + threadIdx.x) >> 5);
  const int lane = threadIdx.x & 31;
  if (wave >= p.tiles_m * p.ngroups) return;
  const int tm = wave % p.tiles_m;
  const int tg = wave / p.tiles_m;
  const int z  = blockIdx.z;
  const int zo = z / p.inner, zi = z - zo * p.inner;
  const __bf16* __restrict__ A  = p.A  + zo * p.sAo + zi * p.sAi;
  const __bf16* __restrict__ Bt = p.Bt + zo * p.sBo + zi * p.sBi;
  const long long coff    = (long long)zo * p.sCo + (long long)zi * p.sCi;
  const long long addoff  = (long long)zo * p.sAddo;
  const long long biasoff = (long long)zo * p.sBiaso;

  const int m0 = tm * 16, nb = tg * 64;
  const int jmax = min(4, (p.N >> 4) - tg * 4);
  const int arow = m0 + (lane & 15);
  const int akh  = (lane & 16) ? 16 : 0;
  const int bcol = lane & 15;
  const int bkh  = (lane & 16) ? 16 : 0;
  const __bf16* __restrict__ ap  = A  + (long long)arow * p.lda + akh;
  const __bf16* __restrict__ bp0 = Bt + (long long)(nb + bcol) * p.ldb + bkh;
  const long long bts = (long long)16 * p.ldb;  // Bt rows per n-tile

  v8f acc0 = {0,0,0,0,0,0,0,0};
  v8f acc1 = acc0, acc2 = acc0, acc3 = acc0;

  if (jmax == 4) {
    // ---- branch-free, software-pipelined hot path ----
    v16bf a  = ldA(ap);
    v16bf b0 = ldB(bp0);
    v16bf b1 = ldB(bp0 + bts);
    v16bf b2 = ldB(bp0 + 2 * bts);
    v16bf b3 = ldB(bp0 + 3 * bts);
    for (int k0 = 32; k0 < p.K; k0 += 32) {
      v16bf an = ldA(ap + k0);
      v16bf c0 = ldB(bp0 + k0);
      v16bf c1 = ldB(bp0 + bts + k0);
      v16bf c2 = ldB(bp0 + 2 * bts + k0);
      v16bf c3 = ldB(bp0 + 3 * bts + k0);
      acc0 = WMMA_BF16(a, b0, acc0);
      acc1 = WMMA_BF16(a, b1, acc1);
      acc2 = WMMA_BF16(a, b2, acc2);
      acc3 = WMMA_BF16(a, b3, acc3);
      a = an; b0 = c0; b1 = c1; b2 = c2; b3 = c3;
    }
    acc0 = WMMA_BF16(a, b0, acc0);
    acc1 = WMMA_BF16(a, b1, acc1);
    acc2 = WMMA_BF16(a, b2, acc2);
    acc3 = WMMA_BF16(a, b3, acc3);
  } else {
    // ---- edge path (only last n-group of N=208 GEMMs) ----
    for (int k0 = 0; k0 < p.K; k0 += 32) {
      v16bf a = ldA(ap + k0);
      acc0 = WMMA_BF16(a, ldB(bp0 + k0), acc0);
      if (jmax > 1) acc1 = WMMA_BF16(a, ldB(bp0 + bts + k0), acc1);
      if (jmax > 2) acc2 = WMMA_BF16(a, ldB(bp0 + 2 * bts + k0), acc2);
      if (jmax > 3) acc3 = WMMA_BF16(a, ldB(bp0 + 3 * bts + k0), acc3);
    }
  }

  const int rb = (lane & 16) ? 8 : 0;
#define EPILOG(J, ACC)                                                        \
  if (jmax > (J)) {                                                           \
    int col = nb + (J) * 16 + (lane & 15);                                    \
    float bsv = p.bias ? p.bias[biasoff + col] : 0.f;                         \
    for (int i = 0; i < 8; ++i) {                                             \
      long long row = m0 + rb + i;                                            \
      float v = ACC[i] * p.alpha + bsv;                                       \
      if (p.Add) v += (float)p.Add[addoff + row * (long long)p.ldadd + col];  \
      if (p.relu && v < 0.f) v = 0.f;                                         \
      long long idx = coff + row * (long long)p.ldc + col;                    \
      if (p.Cf) p.Cf[idx] = v;                                                \
      if (p.Cb) p.Cb[idx] = (__bf16)v;                                        \
    }                                                                         \
  }
  EPILOG(0, acc0) EPILOG(1, acc1) EPILOG(2, acc2) EPILOG(3, acc3)
#undef EPILOG
}

// ---------------- helper kernels ----------------
__global__ void k_zero_b16(__bf16* p, long long n) {
  long long i = (long long)blockIdx.x * 256 + threadIdx.x;
  if (i < n) p[i] = (__bf16)0.f;
}
__global__ void k_zero_f32(float* p, long long n) {
  long long i = (long long)blockIdx.x * 256 + threadIdx.x;
  if (i < n) p[i] = 0.f;
}

// f32 [rows,kin] -> bf16 [rows,kout] (zero pad K)
__global__ void k_conv_pad(const float* in, __bf16* out, int rows, int kin, int kout) {
  long long i = (long long)blockIdx.x * 256 + threadIdx.x;
  if (i >= (long long)rows * kout) return;
  int k = (int)(i % kout); long long r = i / kout;
  out[i] = (k < kin) ? (__bf16)in[r * kin + k] : (__bf16)0.f;
}
// out[n*kout+k] = (n<nvalid && k<kin) ? in[k*in_ld + n] : 0   (transpose + pad)
__global__ void k_transpose_pad(const float* in, __bf16* out, int nrows_out, int nvalid,
                                int kin, int kout, int in_ld) {
  long long i = (long long)blockIdx.x * 256 + threadIdx.x;
  if (i >= (long long)nrows_out * kout) return;
  int k = (int)(i % kout); int n = (int)(i / kout);
  float v = (n < nvalid && k < kin) ? in[(long long)k * in_ld + n] : 0.f;
  out[i] = (__bf16)v;
}
// relu + f32->bf16
__global__ void k_relu_conv(const float* in, __bf16* out, long long n) {
  long long i = (long long)blockIdx.x * 256 + threadIdx.x;
  if (i >= n) return;
  float v = in[i];
  out[i] = (__bf16)(v > 0.f ? v : 0.f);
}
// XW1^T: xw1t[b][n][k] = sum_f node[b][k][f]*gc1[f][n]   (k<200 else 0)
__global__ void k_xw1t(const float* nodef, const float* gc1, __bf16* xw1t) {
  long long i = (long long)blockIdx.x * 256 + threadIdx.x;
  if (i >= (long long)BB * HH * KP) return;
  int k = (int)(i % KP); long long r = i / KP;
  int n = (int)(r % HH); int b = (int)(r / HH);
  float v = 0.f;
  if (k < NN) {
    const float* x = nodef + ((long long)b * NN + k) * FF;
    for (int f = 0; f < FF; ++f) v += x[f] * gc1[f * HH + n];
  }
  xw1t[i] = (__bf16)v;
}
// adj -> bf16 [B,208,224] zero padded
__global__ void k_adjfill(const float* adj, __bf16* out) {
  long long i = (long long)blockIdx.x * 256 + threadIdx.x;
  if (i >= (long long)BB * NP * KP) return;
  int k = (int)(i % KP); long long r = i / KP;
  int m = (int)(r % NP); int b = (int)(r / NP);
  float v = (m < NN && k < NN) ? adj[((long long)b * NN + m) * NN + k] : 0.f;
  out[i] = (__bf16)v;
}
// (h1@gc2)^T: out[b][n][k] = C[b][k][n]  (k<200 else 0)
__global__ void k_h2at(const float* C, __bf16* out) {
  long long i = (long long)blockIdx.x * 256 + threadIdx.x;
  if (i >= (long long)BB * HH * KP) return;
  int k = (int)(i % KP); long long r = i / KP;
  int n = (int)(r % HH); int b = (int)(r / HH);
  float v = (k < NN) ? C[((long long)b * NP + k) * HH + n] : 0.f;
  out[i] = (__bf16)v;
}
// V^T per (b,h): vt[(b*4+h)][d][k] = qkv[b][k][1024+h*128+d]
__global__ void k_vt(const __bf16* qkv, __bf16* vt) {
  long long i = (long long)blockIdx.x * 256 + threadIdx.x;
  if (i >= (long long)BB * 4 * 128 * KP) return;
  int k = (int)(i % KP); long long r = i / KP;
  int d = (int)(r % 128); long long r2 = r / 128;
  int h = (int)(r2 % 4); int b = (int)(r2 / 4);
  __bf16 v = (__bf16)0.f;
  if (k < NN) v = qkv[((long long)b * NP + k) * 1536 + 1024 + h * 128 + d];
  vt[i] = v;
}
// wave-per-row softmax over 200 keys; pads zeroed
__global__ void k_softmax(const float* S, __bf16* P) {
  long long w = ((long long)blockIdx.x * 256 + threadIdx.x) >> 5;
  int lane = threadIdx.x & 31;
  if (w >= (long long)BB * 4 * NP) return;
  int m = (int)(w % NP);
  const float* s = S + w * NP;
  __bf16* pr = P + w * KP;
  if (m >= NN) { for (int k = lane; k < KP; k += 32) pr[k] = (__bf16)0.f; return; }
  float mx = -3.4e38f;
  for (int k = lane; k < NN; k += 32) { float v = s[k]; mx = v > mx ? v : mx; }
  for (int d = 16; d > 0; d >>= 1) { float o = __shfl_xor(mx, d); mx = o > mx ? o : mx; }
  float sum = 0.f;
  for (int k = lane; k < NN; k += 32) sum += __expf(s[k] - mx);
  for (int d = 16; d > 0; d >>= 1) sum += __shfl_xor(sum, d);
  float inv = 1.f / sum;
  for (int k = lane; k < KP; k += 32)
    pr[k] = (k < NN) ? (__bf16)(__expf(s[k] - mx) * inv) : (__bf16)0.f;
}
// mean over nodes
__global__ void k_gf(const float* O, float* gf) {
  int i = blockIdx.x * 256 + threadIdx.x;
  if (i >= BB * HH) return;
  int j = i % HH, b = i / HH;
  const float* p = O + (long long)b * NP * HH + j;
  float s = 0.f;
  for (int n = 0; n < NN; ++n) s += p[(long long)n * HH];
  gf[i] = s * (1.f / NN);
}
// build [feat || gf] bf16, K padded to 544
struct FeatP { const float* f[5]; int fd[5]; };
__global__ void k_xcat(FeatP fp, const float* gf, __bf16* xcat) {
  long long i = (long long)blockIdx.x * 256 + threadIdx.x;
  if (i >= (long long)5 * BT * KIN0) return;
  int k = (int)(i % KIN0); long long r = i / KIN0;
  int row = (int)(r % BT); int br = (int)(r / BT);
  int b = row / TT;
  int fd = fp.fd[br];
  float v = 0.f;
  if (k < fd) v = fp.f[br][(long long)row * fd + k];
  else if (k < fd + HH) v = gf[b * HH + (k - fd)];
  xcat[i] = (__bf16)v;
}
// bias sums bih+bhh per (branch,layer)
struct BiasP { const float* bih[10]; const float* bhh[10]; };
__global__ void k_bsum(BiasP bp, float* out) {
  int i = blockIdx.x * 256 + threadIdx.x;
  if (i >= 10 * G4) return;
  int j = i % G4, r = i / G4;
  out[i] = bp.bih[r][j] + bp.bhh[r][j];
}
// gather 8 bias vectors with zero pad
struct P8 { const float* p[8]; int n[8]; };
__global__ void k_gather8(P8 g, float* out, int stride) {
  int i = blockIdx.x * 256 + threadIdx.x;
  if (i >= 8 * stride) return;
  int j = i % stride, h = i / stride;
  out[i] = (j < g.n[h]) ? g.p[h][j] : 0.f;
}
DEV float sigf(float x) { return 1.f / (1.f + __expf(-x)); }
// fused LSTM gates (PyTorch order i,f,g,o), z-batched over 5 branches
__global__ void k_gates(const float* gbuf, float* cst, __bf16* hb, __bf16* h1seq,
                        int t, int layer) {
  long long i = (long long)blockIdx.x * 256 + threadIdx.x;
  if (i >= (long long)5 * BB * HH) return;
  int j = (int)(i % HH); long long r = i / HH;
  int b = (int)(r % BB); int br = (int)(r / BB);
  const float* g = gbuf + ((long long)br * BB + b) * G4;
  float gi = sigf(g[j]);
  float gF = sigf(g[j + 512]);
  float gg = tanhf(g[j + 1024]);
  float go = sigf(g[j + 1536]);
  float c = gF * cst[i] + gi * gg;
  cst[i] = c;
  float h = go * tanhf(c);
  hb[i] = (__bf16)h;
  if (layer == 0)
    h1seq[((long long)br * BT + b * TT + t) * HH + j] = (__bf16)h;
}
// arrange per-head hidden state: heads 0..7 <- branches {0,0,1,1,2,2,3,4}
__global__ void k_hhead(const __bf16* hb, __bf16* out) {
  const int map[8] = {0, 0, 1, 1, 2, 2, 3, 4};
  long long i = (long long)blockIdx.x * 256 + threadIdx.x;
  if (i >= (long long)8 * BB * HH) return;
  int j = (int)(i % (BB * HH)); int h = (int)(i / (BB * HH));
  out[i] = hb[(long long)map[h] * BB * HH + j];
}
// final activations + scatter into d_out
__global__ void k_heads_out(const float* ho, float* out) {
  int i = blockIdx.x * 256 + threadIdx.x;
  if (i >= 8 * BB * 64) return;
  int c = i % 64, r = i / 64;
  int b = r % BB, h = r / BB;
  int outd = (h < 4) ? 60 : 30;
  if (c >= outd) return;
  float v = ho[i];
  if (h == 1 || h == 3 || h == 5) v = (v > 20.f) ? v : log1pf(__expf(v));  // softplus
  if (h == 6 || h == 7) v = 1.f / (1.f + __expf(-v));                      // sigmoid
  const long long base[8] = {0, 7680, 15360, 23040, 30720, 34560, 38400, 42240};
  out[base[h] + (long long)b * outd + c] = v;
}

// ---------------- host side ----------------
static GemmP GP() { GemmP p{}; p.inner = 1; p.alpha = 1.f; return p; }

static void run_gemm(hipStream_t s, GemmP p, int zc) {
  p.tiles_m = p.M / 16;
  p.ngroups = (p.N / 16 + 3) / 4;
  int waves = p.tiles_m * p.ngroups;
  dim3 grid((unsigned)((waves + 7) / 8), 1, (unsigned)zc);
  wmma_gemm_bf16<<<grid, 256, 0, s>>>(p);
}
static inline dim3 elg(long long n) { return dim3((unsigned)((n + 255) / 256)); }

extern "C" void kernel_launch(void* const* d_in, const int* in_sizes, int n_in,
                              void* d_out, int out_size, void* d_ws, size_t ws_size,
                              hipStream_t stream) {
  (void)in_sizes; (void)n_in; (void)out_size; (void)ws_size;
  const float* IN[96];
  for (int i = 0; i < n_in && i < 96; ++i) IN[i] = (const float*)d_in[i];

  // ---- workspace bump allocator ----
  char* wsb = (char*)d_ws; size_t off = 0;
  auto alloc = [&](size_t bytes) -> void* {
    size_t a = (off + 255) & ~(size_t)255; off = a + bytes; return wsb + a;
  };
  __bf16* Wih0b = (__bf16*)alloc((size_t)5 * G4 * KIN0 * 2);
  __bf16* Whh0b = (__bf16*)alloc((size_t)5 * G4 * HH * 2);
  __bf16* Wih1b = (__bf16*)alloc((size_t)5 * G4 * HH * 2);
  __bf16* Whh1b = (__bf16*)alloc((size_t)5 * G4 * HH * 2);
  __bf16* W2t   = (__bf16*)alloc((size_t)HH * HH * 2);
  __bf16* AinW  = (__bf16*)alloc((size_t)1536 * HH * 2);
  __bf16* AoutW = (__bf16*)alloc((size_t)HH * HH * 2);
  __bf16* W1t   = (__bf16*)alloc((size_t)8 * HH * HH * 2);
  __bf16* W2tH  = (__bf16*)alloc((size_t)8 * 64 * HH * 2);
  float*  BSUM  = (float*)alloc((size_t)10 * G4 * 4);
  float*  B1A   = (float*)alloc((size_t)8 * HH * 4);
  float*  B2A   = (float*)alloc((size_t)8 * 64 * 4);
  __bf16* XW1T  = (__bf16*)alloc((size_t)BB * HH * KP * 2);
  __bf16* ADJ   = (__bf16*)alloc((size_t)BB * NP * KP * 2);
  __bf16* H1B   = (__bf16*)alloc((size_t)MBN * HH * 2);
  __bf16* H2B   = (__bf16*)alloc((size_t)MBN * HH * 2);
  __bf16* H2AT  = (__bf16*)alloc((size_t)BB * HH * KP * 2);
  __bf16* QKV   = (__bf16*)alloc((size_t)MBN * 1536 * 2);
  __bf16* PATT  = (__bf16*)alloc((size_t)BB * 4 * NP * KP * 2);
  __bf16* VT    = (__bf16*)alloc((size_t)BB * 4 * 128 * KP * 2);
  __bf16* OB    = (__bf16*)alloc((size_t)MBN * HH * 2);
  float*  CBUF  = (float*)alloc((size_t)BB * 4 * NP * NP * 4);  // reused f32 scratch
  float*  GF    = (float*)alloc((size_t)BB * HH * 4);
  __bf16* XCAT  = (__bf16*)alloc((size_t)5 * BT * KIN0 * 2);
  __bf16* XG    = (__bf16*)alloc((size_t)5 * BT * G4 * 2);      // reused layer0/layer1
  __bf16* H1SEQ = (__bf16*)alloc((size_t)5 * BT * HH * 2);
  __bf16* HB    = (__bf16*)alloc((size_t)5 * BB * HH * 2);
  float*  CST   = (float*)alloc((size_t)5 * BB * HH * 4);
  float*  GBUF  = (float*)alloc((size_t)5 * BB * G4 * 4);
  __bf16* HHEAD = (__bf16*)alloc((size_t)8 * BB * HH * 2);
  __bf16* HID   = (__bf16*)alloc((size_t)8 * BB * HH * 2);
  float*  HOUT  = (float*)alloc((size_t)8 * BB * 64 * 4);

  // ---- 1) weight conversion (f32 -> bf16 [N,K], padded) ----
  for (int br = 0; br < 5; ++br) {
    int base = 13 + br * 8;
    int ind = (br < 2) ? 514 : 513;
    k_conv_pad<<<elg((long long)G4 * KIN0), 256, 0, stream>>>(IN[base + 0], Wih0b + (size_t)br * G4 * KIN0, G4, ind, KIN0);
    k_conv_pad<<<elg((long long)G4 * HH), 256, 0, stream>>>(IN[base + 1], Whh0b + (size_t)br * G4 * HH, G4, HH, HH);
    k_conv_pad<<<elg((long long)G4 * HH), 256, 0, stream>>>(IN[base + 4], Wih1b + (size_t)br * G4 * HH, G4, HH, HH);
    k_conv_pad<<<elg((long long)G4 * HH), 256, 0, stream>>>(IN[base + 5], Whh1b + (size_t)br * G4 * HH, G4, HH, HH);
  }
  k_transpose_pad<<<elg((long long)HH * HH), 256, 0, stream>>>(IN[8], W2t, HH, HH, HH, HH, HH);
  k_conv_pad<<<elg((long long)1536 * HH), 256, 0, stream>>>(IN[9], AinW, 1536, HH, HH);
  k_conv_pad<<<elg((long long)HH * HH), 256, 0, stream>>>(IN[11], AoutW, HH, HH, HH);
  for (int h = 0; h < 8; ++h) {
    int hb = 53 + h * 4, outd = (h < 4) ? 60 : 30;
    k_transpose_pad<<<elg((long long)HH * HH), 256, 0, stream>>>(IN[hb + 0], W1t + (size_t)h * HH * HH, HH, HH, HH, HH, HH);
    k_transpose_pad<<<elg((long long)64 * HH), 256, 0, stream>>>(IN[hb + 2], W2tH + (size_t)h * 64 * HH, 64, outd, HH, HH, outd);
  }
  { BiasP bp;
    for (int br = 0; br < 5; ++br) {
      int base = 13 + br * 8;
      bp.bih[br * 2 + 0] = IN[base + 2]; bp.bhh[br * 2 + 0] = IN[base + 3];
      bp.bih[br * 2 + 1] = IN[base + 6]; bp.bhh[br * 2 + 1] = IN[base + 7];
    }
    k_bsum<<<elg(10 * G4), 256, 0, stream>>>(bp, BSUM);
  }
  { P8 g; for (int h = 0; h < 8; ++h) { g.p[h] = IN[53 + h * 4 + 1]; g.n[h] = HH; }
    k_gather8<<<elg(8 * HH), 256, 0, stream>>>(g, B1A, HH); }
  { P8 g; for (int h = 0; h < 8; ++h) { g.p[h] = IN[53 + h * 4 + 3]; g.n[h] = (h < 4) ? 60 : 30; }
    k_gather8<<<elg(8 * 64), 256, 0, stream>>>(g, B2A, 64); }

  // ---- 2) GCN ----
  k_xw1t<<<elg((long long)BB * HH * KP), 256, 0, stream>>>(IN[5], IN[7], XW1T);
  k_adjfill<<<elg((long long)BB * NP * KP), 256, 0, stream>>>(IN[6], ADJ);
  { GemmP p = GP();                                    // adj @ (X@W1) -> CBUF
    p.A = ADJ; p.lda = KP; p.sAo = (long long)NP * KP;
    p.Bt = XW1T; p.ldb = KP; p.sBo = (long long)HH * KP;
    p.Cf = CBUF; p.ldc = HH; p.sCo = (long long)NP * HH;
    p.M = NP; p.N = HH; p.K = KP; run_gemm(stream, p, BB); }
  k_relu_conv<<<elg((long long)MBN * HH), 256, 0, stream>>>(CBUF, H1B, (long long)MBN * HH);
  { GemmP p = GP();                                    // h1 @ gc2 -> CBUF
    p.A = H1B; p.lda = HH; p.Bt = W2t; p.ldb = HH;
    p.Cf = CBUF; p.ldc = HH; p.M = MBN; p.N = HH; p.K = HH; run_gemm(stream, p, 1); }
  k_h2at<<<elg((long long)BB * HH * KP), 256, 0, stream>>>(CBUF, H2AT);
  { GemmP p = GP();                                    // adj @ (h1@gc2) -> CBUF
    p.A = ADJ; p.lda = KP; p.sAo = (long long)NP * KP;
    p.Bt = H2AT; p.ldb = KP; p.sBo = (long long)HH * KP;
    p.Cf = CBUF; p.ldc = HH; p.sCo = (long long)NP * HH;
    p.M = NP; p.N = HH; p.K = KP; run_gemm(stream, p, BB); }
  k_relu_conv<<<elg((long long)MBN * HH), 256, 0, stream>>>(CBUF, H2B, (long long)MBN * HH);

  // ---- 3) attention ----
  { GemmP p = GP();                                    // qkv = h @ Win^T + b
    p.A = H2B; p.lda = HH; p.Bt = AinW; p.ldb = HH; p.bias = IN[10];
    p.Cb = QKV; p.ldc = 1536; p.M = MBN; p.N = 1536; p.K = HH; run_gemm(stream, p, 1); }
  { GemmP p = GP();                                    // scores = q @ k^T / sqrt(128)
    p.A = QKV; p.lda = 1536; p.sAo = (long long)NP * 1536; p.sAi = 128; p.inner = 4;
    p.Bt = QKV + 512; p.ldb = 1536; p.sBo = (long long)NP * 1536; p.sBi = 128;
    p.Cf = CBUF; p.ldc = NP; p.sCo = (long long)4 * NP * NP; p.sCi = (long long)NP * NP;
    p.M = NP; p.N = NP; p.K = 128; p.alpha = 0.08838834764831845f;
    run_gemm(stream, p, BB * 4); }
  k_softmax<<<elg((long long)BB * 4 * NP * 32), 256, 0, stream>>>(CBUF, PATT);
  k_vt<<<elg((long long)BB * 4 * 128 * KP), 256, 0, stream>>>(QKV, VT);
  { GemmP p = GP();                                    // o = P @ V  (head-concat on store)
    p.A = PATT; p.lda = KP; p.sAo = (long long)4 * NP * KP; p.sAi = (long long)NP * KP; p.inner = 4;
    p.Bt = VT; p.ldb = KP; p.sBo = (long long)4 * 128 * KP; p.sBi = (long long)128 * KP;
    p.Cb = OB; p.ldc = HH; p.sCo = (long long)NP * HH; p.sCi = 128;
    p.M = NP; p.N = 128; p.K = KP; run_gemm(stream, p, BB * 4); }
  { GemmP p = GP();                                    // out proj
    p.A = OB; p.lda = HH; p.Bt = AoutW; p.ldb = HH; p.bias = IN[12];
    p.Cf = CBUF; p.ldc = HH; p.M = MBN; p.N = HH; p.K = HH; run_gemm(stream, p, 1); }
  k_gf<<<elg(BB * HH), 256, 0, stream>>>(CBUF, GF);

  // ---- 4) LSTM branches (z-batched over 5) ----
  { FeatP fp; const int fds[5] = {2, 2, 1, 1, 1};
    for (int br = 0; br < 5; ++br) { fp.f[br] = IN[br]; fp.fd[br] = fds[br]; }
    k_xcat<<<elg((long long)5 * BT * KIN0), 256, 0, stream>>>(fp, GF, XCAT); }
  { GemmP p = GP();                                    // xg0 = xcat @ Wih0^T + (bih0+bhh0)
    p.A = XCAT; p.lda = KIN0; p.sAo = (long long)BT * KIN0;
    p.Bt = Wih0b; p.ldb = KIN0; p.sBo = (long long)G4 * KIN0;
    p.bias = BSUM; p.sBiaso = 2 * G4;
    p.Cb = XG; p.ldc = G4; p.sCo = (long long)BT * G4;
    p.M = BT; p.N = G4; p.K = KIN0; run_gemm(stream, p, 5); }

  for (int layer = 0; layer < 2; ++layer) {
    k_zero_b16<<<elg((long long)5 * BB * HH), 256, 0, stream>>>(HB, (long long)5 * BB * HH);
    k_zero_f32<<<elg((long long)5 * BB * HH), 256, 0, stream>>>(CST, (long long)5 * BB * HH);
    const __bf16* Whh = (layer == 0) ? Whh0b : Whh1b;
    for (int t = 0; t < TT; ++t) {
      GemmP p = GP();                                  // g = xg_t + h @ Whh^T
      p.A = HB; p.lda = HH; p.sAo = (long long)BB * HH;
      p.Bt = Whh; p.ldb = HH; p.sBo = (long long)G4 * HH;
      p.Add = XG + (long long)t * G4; p.ldadd = TT * G4; p.sAddo = (long long)BT * G4;
      p.Cf = GBUF; p.ldc = G4; p.sCo = (long long)BB * G4;
      p.M = BB; p.N = G4; p.K = HH; run_gemm(stream, p, 5);
      k_gates<<<elg((long long)5 * BB * HH), 256, 0, stream>>>(GBUF, CST, HB, H1SEQ, t, layer);
    }
    if (layer == 0) {                                  // xg1 = h1seq @ Wih1^T + (bih1+bhh1)
      GemmP p = GP();
      p.A = H1SEQ; p.lda = HH; p.sAo = (long long)BT * HH;
      p.Bt = Wih1b; p.ldb = HH; p.sBo = (long long)G4 * HH;
      p.bias = BSUM + G4; p.sBiaso = 2 * G4;
      p.Cb = XG; p.ldc = G4; p.sCo = (long long)BT * G4;
      p.M = BT; p.N = G4; p.K = HH; run_gemm(stream, p, 5);
    }
  }

  // ---- 5) heads ----
  k_hhead<<<elg((long long)8 * BB * HH), 256, 0, stream>>>(HB, HHEAD);
  { GemmP p = GP();                                    // hid = relu(h @ w1^T + b1)
    p.A = HHEAD; p.lda = HH; p.sAo = (long long)BB * HH;
    p.Bt = W1t; p.ldb = HH; p.sBo = (long long)HH * HH;
    p.bias = B1A; p.sBiaso = HH;
    p.Cb = HID; p.ldc = HH; p.sCo = (long long)BB * HH; p.relu = 1;
    p.M = BB; p.N = HH; p.K = HH; run_gemm(stream, p, 8); }
  { GemmP p = GP();                                    // out = hid @ w2^T + b2
    p.A = HID; p.lda = HH; p.sAo = (long long)BB * HH;
    p.Bt = W2tH; p.ldb = HH; p.sBo = (long long)64 * HH;
    p.bias = B2A; p.sBiaso = 64;
    p.Cf = HOUT; p.ldc = 64; p.sCo = (long long)BB * 64;
    p.M = BB; p.N = 64; p.K = HH; run_gemm(stream, p, 8); }
  k_heads_out<<<elg(8 * BB * 64), 256, 0, stream>>>(HOUT, (float*)d_out);
}
